// Test_57277683859886
// MI455X (gfx1250) — compile-verified, matrix operand order checked by reading, not run on hardware
//
#include <hip/hip_runtime.h>

// ---------------------------------------------------------------------------
// Binarized 3x3 conv (sign(x) * sign(W), pad 1) as implicit GEMM on CDNA5 WMMA.
//   A (16x32 f16) = binarized weights (M = cout tile, K = cin = 32)
//   B (32x16 f16) = binarized input   (K = cin, N = 16 output columns)
//   D (16x16 f32) accumulators; 9 taps x 2 M-tiles = 18 v_wmma per out block.
// Memory-bound (~302 MB @ 23.3 TB/s): single HBM pass, 4-slot LDS row ring,
// software-pipelined so global loads for row h+2 fly under the WMMAs of row h.
// ---------------------------------------------------------------------------

typedef _Float16 v16h __attribute__((ext_vector_type(16)));
typedef _Float16 v8h  __attribute__((ext_vector_type(8)));
typedef float    v8f  __attribute__((ext_vector_type(8)));

#define N_BATCH 8
#define C_IN    32
#define C_OUT   32
#define HH      384
#define WW      384
#define WTILE   128             // output columns per workgroup (8 waves x 16)
#define HCHUNK  24              // output rows per workgroup
#define LCOLS   (WTILE + 2)     // staged columns incl. 1-col halo each side
#define NSTAGE  ((C_IN * LCOLS + 255) / 256)   // 17 staged elems per thread
#define SLOT(r) (((r) + 4) & 3)                // 4-deep row ring

__device__ __forceinline__ _Float16 sgnh(float v) {
    return (v > 0.f) ? (_Float16)1.f : ((v < 0.f) ? (_Float16)-1.f : (_Float16)0.f);
}

__global__ __launch_bounds__(256)
void binconv_wmma(const float* __restrict__ x, const float* __restrict__ w,
                  float* __restrict__ out)
{
    // input ring: [slot][col][cin]: a lane's 8-channel K-run is one ds_load_b128
    __shared__ _Float16 xs[4][LCOLS][C_IN];                 // 33.3 KB
    // weights pre-packed in per-lane WMMA-A fragment order (ISA 7.12.2):
    // wl[tap][mtile][lane][pos], pos=j*2+p,
    // K(cin) = (j<4 ? 0:16) + (lane>=16 ? 8:0) + (j&3)*2 + p
    __shared__ _Float16 wl[9][2][32][16];                   // 18.4 KB

    const int tid   = threadIdx.x;
    const int lane  = tid & 31;       // wave32
    const int wave  = tid >> 5;       // 0..7
    const int nloc  = lane & 15;
    const int half  = lane >> 4;
    const int kbase = half * 8;

    const int w0 = blockIdx.x * WTILE;
    const int h0 = blockIdx.y * HCHUNK;
    const int nb = blockIdx.z;

    // ---- one-time: binarize + pack weights into WMMA-A fragment layout ----
    for (int idx = tid; idx < 9 * 2 * 32 * 16; idx += 256) {
        const int pos = idx & 15;
        const int l   = (idx >> 4) & 31;
        const int q   = (idx >> 9) & 1;
        const int t   = idx >> 10;
        const int j = pos >> 1, p = pos & 1;
        const int k = ((j < 4) ? 0 : 16) + ((l >> 4) << 3) + ((j & 3) << 1) + p;
        const int o = q * 16 + (l & 15);
        wl[t][q][l][pos] = sgnh(w[((o * C_IN + k) * 3 + (t / 3)) * 3 + (t % 3)]);
    }

    // ---- hoisted row-invariant staging addresses (per thread, 17 elems) ----
    int goff[NSTAGE];   // global elem offset sans row term; -1 = padded/out
    int lofs[NSTAGE];   // f16 offset within one ring slot; -1 = no element
    #pragma unroll
    for (int u = 0; u < NSTAGE; ++u) {
        const int idx = tid + u * 256;
        if (idx < C_IN * LCOLS) {
            const int ch  = idx / LCOLS;
            const int col = idx - ch * LCOLS;
            const int gw  = w0 + col - 1;
            lofs[u] = col * C_IN + ch;
            goff[u] = (gw >= 0 && gw < WW) ? ((nb * C_IN + ch) * HH * WW + gw) : -1;
        } else { lofs[u] = -1; goff[u] = -1; }
    }

    // ---- combined load+store staging (priming only) ----
    auto stage_full = [&](int row) {
        _Float16* dst = &xs[SLOT(row)][0][0];
        const bool rowok = (row >= 0) && (row < HH);
        #pragma unroll
        for (int u = 0; u < NSTAGE; ++u)
            if (lofs[u] >= 0) {
                const float v = (rowok && goff[u] >= 0) ? x[goff[u] + row * WW] : 0.f;
                dst[lofs[u]] = sgnh(v);
            }
    };
    stage_full(h0 - 1);
    stage_full(h0);
    stage_full(h0 + 1);
    __syncthreads();

    for (int h = h0; h < h0 + HCHUNK; ++h) {
        // phase 1: issue global loads for row h+2 (land during the WMMAs)
        const int lrow = h + 2;
        const bool rowok = lrow < HH;
        float r[NSTAGE];
        #pragma unroll
        for (int u = 0; u < NSTAGE; ++u)
            r[u] = (rowok && goff[u] >= 0) ? x[goff[u] + lrow * WW] : 0.f;
        if (tid < C_IN && (h + 3) < HH)        // warm L2 two rows ahead
            __builtin_prefetch(&x[(nb * C_IN + tid) * HH * WW + (h + 3) * WW + w0], 0, 1);

        // phase 2: 18 WMMAs for output row h (slots h-1..h+1; slot h+2 free)
        v8f acc0 = {};                          // cout 0..15
        v8f acc1 = {};                          // cout 16..31
        const int rings[3] = { SLOT(h - 1), SLOT(h), SLOT(h + 1) };
        #pragma unroll
        for (int kh = 0; kh < 3; ++kh) {
            const int ring = rings[kh];
            #pragma unroll
            for (int kw = 0; kw < 3; ++kw) {
                const int col = wave * 16 + nloc + kw;
                union Frag { v16h v; struct { v8h lo, hi; } s; } B, A0, A1;
                B.s.lo  = *(const v8h*)&xs[ring][col][kbase];
                B.s.hi  = *(const v8h*)&xs[ring][col][kbase + 16];
                const int t = kh * 3 + kw;
                A0.s.lo = *(const v8h*)&wl[t][0][lane][0];
                A0.s.hi = *(const v8h*)&wl[t][0][lane][8];
                A1.s.lo = *(const v8h*)&wl[t][1][lane][0];
                A1.s.hi = *(const v8h*)&wl[t][1][lane][8];
                acc0 = __builtin_amdgcn_wmma_f32_16x16x32_f16(
                           false, A0.v, false, B.v, (short)0, acc0, false, false);
                acc1 = __builtin_amdgcn_wmma_f32_16x16x32_f16(
                           false, A1.v, false, B.v, (short)0, acc1, false, false);
            }
        }

        // phase 3: binarize staged row into slot (h+2)&3 (no reader until the
        // barrier: end-of-iteration barrier bounds wave skew to one iteration)
        {
            _Float16* dst = &xs[SLOT(lrow)][0][0];
            #pragma unroll
            for (int u = 0; u < NSTAGE; ++u)
                if (lofs[u] >= 0) dst[lofs[u]] = sgnh(r[u]);
        }

        // phase 4: write outputs (D layout: vgpr v, lane-half -> M = v + 8*half)
        const int  wcol = w0 + wave * 16 + nloc;
        const long base = ((long)nb * C_OUT) * (HH * WW) + (long)h * WW + wcol;
        #pragma unroll
        for (int v = 0; v < 8; ++v) {
            const int o0 = v + 8 * half;
            out[base + (long)o0        * (HH * WW)] = acc0[v];
            out[base + (long)(o0 + 16) * (HH * WW)] = acc1[v];
        }

        __syncthreads();   // single barrier per row
    }
}

extern "C" void kernel_launch(void* const* d_in, const int* in_sizes, int n_in,
                              void* d_out, int out_size, void* d_ws, size_t ws_size,
                              hipStream_t stream) {
    const float* x = (const float*)d_in[0];   // (8,32,384,384) f32
    const float* w = (const float*)d_in[1];   // (32,32,3,3)    f32
    float* o = (float*)d_out;                 // (8,32,384,384) f32
    dim3 grid(WW / WTILE, HH / HCHUNK, N_BATCH);   // 3 x 16 x 8 = 384 WGs
    binconv_wmma<<<grid, 256, 0, stream>>>(x, w, o);
}
